// MultiHeadAttention_74217034875165
// MI455X (gfx1250) — matrix operand order repaired and checked by reading, not compiled, not run on hardware
//
#include <hip/hip_runtime.h>

// MultiHeadAttention (masked) for MI455X / gfx1250, wave32 + WMMA f16->f32.
// B=16, N=1024, E=128, H=8, KD=16.
// Memory-bound on the 64MB int32 adjacency stream (~3.4us roofline at 23.3TB/s);
// all matmuls via v_wmma_f32_16x16x32_f16.

#define B_  16
#define N_  1024
#define E_  128
#define H_  8
#define KD_ 16

typedef __attribute__((ext_vector_type(16))) _Float16 v16h;
typedef __attribute__((ext_vector_type(8)))  _Float16 v8h;
typedef __attribute__((ext_vector_type(8)))  float    v8f;

// Build a 16-half fragment from two contiguous 8-half (16B) chunks.
static __device__ __forceinline__ v16h frag_from2(const _Float16* p0, const _Float16* p1) {
    v8h lo = *(const v8h*)p0;
    v8h hi = *(const v8h*)p1;
    v16h a;
#pragma unroll
    for (int i = 0; i < 8; ++i) { a[i] = lo[i]; a[i + 8] = hi[i]; }
    return a;
}

// ---------------------------------------------------------------------------
// Kernel 0: precision/layout prep.
//  WqT/WkT/WvT : column-major [c=h*16+k][e] f16   (B-fragment friendly)
//  WoT         : column-major [e][rr=h*16+v] f16
//  h_h         : f16 copy of h [B*N][E]
// ---------------------------------------------------------------------------
__global__ void prep_kernel(const float* __restrict__ h,
                            const float* __restrict__ Wq,
                            const float* __restrict__ Wk,
                            const float* __restrict__ Wv,
                            const float* __restrict__ Wo,
                            _Float16* __restrict__ WqT,
                            _Float16* __restrict__ WkT,
                            _Float16* __restrict__ WvT,
                            _Float16* __restrict__ WoT,
                            _Float16* __restrict__ h_h) {
    const long idx = (long)blockIdx.x * blockDim.x + threadIdx.x;
    const int  WE  = H_ * E_ * KD_;           // 16384 elements per weight tensor
    if (idx < 3L * WE) {
        int w = (int)(idx / WE), r = (int)(idx % WE);
        int c = r / E_, e = r % E_;           // c = h*16+k
        int hh = c / KD_, k = c % KD_;
        const float* src = (w == 0) ? Wq : ((w == 1) ? Wk : Wv);
        _Float16*    dst = (w == 0) ? WqT : ((w == 1) ? WkT : WvT);
        dst[c * E_ + e] = (_Float16)src[hh * (E_ * KD_) + e * KD_ + k];
    } else if (idx < 4L * WE) {
        int r  = (int)(idx - 3L * WE);
        int e  = r / (H_ * KD_), rr = r % (H_ * KD_);
        WoT[e * (H_ * KD_) + rr] = (_Float16)Wo[rr * E_ + e];  // Wo is already [rr][e]
    } else {
        long i2 = idx - 4L * WE;
        if (i2 < (long)B_ * N_ * E_) h_h[i2] = (_Float16)h[i2];
    }
}

// ---------------------------------------------------------------------------
// Kernel 1: QKV projection.  grid = (B*N/16, 3), block = 32 (one wave).
// [16 x 128] = h_tile[16 x 128] @ W[128 x 128]; 4 k-steps x 8 column tiles.
// Q,K stored [B,H,N,16]; V stored transposed [B,H,16,N] so the attention
// B-fragment for V is two contiguous 16B loads per lane.
// ---------------------------------------------------------------------------
__global__ void qkv_kernel(const _Float16* __restrict__ h_h,
                           const _Float16* __restrict__ WqT,
                           const _Float16* __restrict__ WkT,
                           const _Float16* __restrict__ WvT,
                           _Float16* __restrict__ Qh,
                           _Float16* __restrict__ Kh,
                           _Float16* __restrict__ Vt) {
    const int l = threadIdx.x, lane16 = l & 15, g = l >> 4;
    const int row0 = blockIdx.x * 16;
    const int sel  = blockIdx.y;
    const _Float16* WT = (sel == 0) ? WqT : ((sel == 1) ? WkT : WvT);

    v8f acc[8];
#pragma unroll
    for (int c = 0; c < 8; ++c) acc[c] = {};

#pragma unroll
    for (int ks = 0; ks < 4; ++ks) {
        // A fragment: lane holds row m=lane16; K = ks*32 + (g*8..g*8+7, 16+g*8..)
        const _Float16* arow = h_h + (long)(row0 + lane16) * E_ + ks * 32 + g * 8;
        v16h a = frag_from2(arow, arow + 16);
#pragma unroll
        for (int c = 0; c < 8; ++c) {
            // B fragment: lane holds col = c*16+lane16; K rows = ks*32 + (g*8..g*8+7, 16+g*8..)
            const _Float16* bcol = WT + (long)(c * 16 + lane16) * E_ + ks * 32 + g * 8;
            v16h bf = frag_from2(bcol, bcol + 16);
            acc[c] = __builtin_amdgcn_wmma_f32_16x16x32_f16(false, a, false, bf,
                                                            (short)0, acc[c], false, false);
        }
    }

    const float qs = (sel == 0) ? 0.25f : 1.0f;  // fold 1/sqrt(KD) into Q
#pragma unroll
    for (int c = 0; c < 8; ++c) {                // column tile c == head c
#pragma unroll
        for (int r = 0; r < 8; ++r) {
            int rowg = row0 + r + 8 * g;
            int b = rowg >> 10, n = rowg & (N_ - 1);
            float v = acc[c][r] * qs;
            if (sel == 2) {
                // Vt[b][head c][kd=lane16][n]
                Vt[((long)(b * H_ + c) * KD_ + lane16) * N_ + n] = (_Float16)v;
            } else {
                _Float16* out = (sel == 0) ? Qh : Kh;   // [b][head c][n][kd]
                out[((long)(b * H_ + c) * N_ + n) * KD_ + lane16] = (_Float16)v;
            }
        }
    }
}

// ---------------------------------------------------------------------------
// Kernel 2: masked flash attention.  grid = (N/16, H, B), block = 32.
// Per 32-key iteration: 2 WMMA (S) + 1 WMMA (P@V), online softmax in fp32.
// ---------------------------------------------------------------------------
__global__ void attn_kernel(const int* __restrict__ adj,
                            const _Float16* __restrict__ Qh,
                            const _Float16* __restrict__ Kh,
                            const _Float16* __restrict__ Vt,
                            _Float16* __restrict__ heads) {
    __shared__ float s_p[16 * 33];               // stride 33: conflict-free transpose
    const int l = threadIdx.x, lane16 = l & 15, g = l >> 4;
    const int qt = blockIdx.x, hd = blockIdx.y, b = blockIdx.z;
    const long bh = (long)(b * H_ + hd) * N_;

    // Q fragment, zero-padded to K=32 (elements 8..15 map to K>=16 -> 0)
    v16h qf;
    {
        const _Float16* qp = Qh + (bh + qt * 16 + lane16) * KD_ + g * 8;
        v8h lo = *(const v8h*)qp;
#pragma unroll
        for (int i = 0; i < 8; ++i) { qf[i] = lo[i]; qf[i + 8] = (_Float16)0.f; }
    }
    // Vt row base for this lane's output column v=lane16
    const _Float16* vtrow = Vt + ((long)(b * H_ + hd) * KD_ + lane16) * N_;

    v8f acc = {};
    float mrow[8], lsum[8];
#pragma unroll
    for (int r = 0; r < 8; ++r) { mrow[r] = -3.0e38f; lsum[r] = 0.f; }

    const int* adjrow = adj + ((long)b * N_ + qt * 16) * N_;

    for (int it = 0; it < N_ / 32; ++it) {
        const int k0 = it * 32;
        if (it + 1 < N_ / 32)
            __builtin_prefetch(adjrow + (long)(8 * g) * N_ + k0 + 32 + lane16, 0, 0);

        // K^T fragments (B layout 32x16): elements 0..7 = kd g*8..g*8+7, elements 8..15 = zero pad
        v16h kf0, kf1;
        {
            const _Float16* kp0 = Kh + (bh + k0 + lane16) * KD_ + g * 8;
            const _Float16* kp1 = Kh + (bh + k0 + 16 + lane16) * KD_ + g * 8;
            v8h lo0 = *(const v8h*)kp0;
            v8h lo1 = *(const v8h*)kp1;
#pragma unroll
            for (int i = 0; i < 8; ++i) { kf0[i] = lo0[i]; kf0[i + 8] = (_Float16)0.f; kf1[i] = lo1[i]; kf1[i + 8] = (_Float16)0.f; }
        }
        v8f z = {};
        v8f s0 = __builtin_amdgcn_wmma_f32_16x16x32_f16(false, qf, false, kf0, (short)0, z, false, false);
        v8f s1 = __builtin_amdgcn_wmma_f32_16x16x32_f16(false, qf, false, kf1, (short)0, z, false, false);

        // mask + online softmax (rows m = r + 8g, cols = lane16 within tile)
#pragma unroll
        for (int r = 0; r < 8; ++r) {
            const int m = r + 8 * g;
            const int a0 = adjrow[(long)m * N_ + k0 + lane16];
            const int a1 = adjrow[(long)m * N_ + k0 + 16 + lane16];
            float v0 = (a0 == 0) ? -3.0e38f : s0[r];
            float v1 = (a1 == 0) ? -3.0e38f : s1[r];
            float mx = fmaxf(v0, v1);
#pragma unroll
            for (int d = 1; d < 16; d <<= 1) mx = fmaxf(mx, __shfl_xor(mx, d, 32));
            float mnew = fmaxf(mrow[r], mx);
            float sc = __expf(mrow[r] - mnew);
            float p0 = (a0 == 0) ? 0.f : __expf(v0 - mnew);
            float p1 = (a1 == 0) ? 0.f : __expf(v1 - mnew);
            float ps = p0 + p1;
#pragma unroll
            for (int d = 1; d < 16; d <<= 1) ps += __shfl_xor(ps, d, 32);
            lsum[r] = lsum[r] * sc + ps;
            mrow[r] = mnew;
            acc[r] *= sc;
            s_p[m * 33 + lane16]      = p0;      // key_local 0..15
            s_p[m * 33 + 16 + lane16] = p1;      // key_local 16..31
        }
        __syncthreads();

        // P as A fragment over 32 keys (LDS transpose C-layout -> A-layout)
        v16h pf;
#pragma unroll
        for (int i = 0; i < 8; ++i) {
            pf[i]     = (_Float16)s_p[lane16 * 33 + g * 8 + i];
            pf[i + 8] = (_Float16)s_p[lane16 * 33 + 16 + g * 8 + i];
        }
        // V as B fragment: element i -> key k0+g*8+i, element i+8 -> key k0+16+g*8+i, col v = lane16.
        // Vt is [kd][key] so this is two contiguous 8-half chunks -> two b128 loads.
        const _Float16* vp = vtrow + k0 + g * 8;
        v16h vf = frag_from2(vp, vp + 16);

        acc = __builtin_amdgcn_wmma_f32_16x16x32_f16(false, pf, false, vf, (short)0, acc, false, false);
        __syncthreads();
    }

    // normalize and store concatenated heads: [B*N][H*KD] f16
#pragma unroll
    for (int r = 0; r < 8; ++r) {
        const int m = r + 8 * g;
        float o = acc[r] / lsum[r];
        heads[((long)b * N_ + qt * 16 + m) * (H_ * KD_) + hd * KD_ + lane16] = (_Float16)o;
    }
}

// ---------------------------------------------------------------------------
// Kernel 3: output projection + residual.  grid = B*N/16, block = 32.
// out[16 x 128] = heads[16 x 128] @ Wo[128 x 128] + h
// ---------------------------------------------------------------------------
__global__ void outproj_kernel(const _Float16* __restrict__ heads,
                               const _Float16* __restrict__ WoT,
                               const float* __restrict__ h,
                               float* __restrict__ out) {
    const int l = threadIdx.x, lane16 = l & 15, g = l >> 4;
    const int row0 = blockIdx.x * 16;
    v8f acc[8];
#pragma unroll
    for (int c = 0; c < 8; ++c) acc[c] = {};

#pragma unroll
    for (int ks = 0; ks < 4; ++ks) {
        const _Float16* arow = heads + (long)(row0 + lane16) * E_ + ks * 32 + g * 8;
        v16h a = frag_from2(arow, arow + 16);
#pragma unroll
        for (int c = 0; c < 8; ++c) {
            const _Float16* bcol = WoT + (long)(c * 16 + lane16) * E_ + ks * 32 + g * 8;
            v16h bf = frag_from2(bcol, bcol + 16);
            acc[c] = __builtin_amdgcn_wmma_f32_16x16x32_f16(false, a, false, bf,
                                                            (short)0, acc[c], false, false);
        }
    }
#pragma unroll
    for (int c = 0; c < 8; ++c) {
#pragma unroll
        for (int r = 0; r < 8; ++r) {
            long idx = (long)(row0 + r + 8 * g) * E_ + c * 16 + lane16;
            out[idx] = acc[c][r] + h[idx];
        }
    }
}

// ---------------------------------------------------------------------------
extern "C" void kernel_launch(void* const* d_in, const int* in_sizes, int n_in,
                              void* d_out, int out_size, void* d_ws, size_t ws_size,
                              hipStream_t stream) {
    (void)in_sizes; (void)n_in; (void)out_size; (void)ws_size;
    const float* h   = (const float*)d_in[0];
    const int*   adj = (const int*)d_in[1];
    const float* Wq  = (const float*)d_in[2];
    const float* Wk  = (const float*)d_in[3];
    const float* Wv  = (const float*)d_in[4];
    const float* Wo  = (const float*)d_in[5];
    float* out = (float*)d_out;

    char* ws = (char*)d_ws;
    const size_t WSZ = (size_t)E_ * (H_ * KD_) * sizeof(_Float16);   // 32 KB
    const size_t HSZ = (size_t)B_ * N_ * E_ * sizeof(_Float16);      // 4 MB
    _Float16* WqT   = (_Float16*)ws; ws += WSZ;
    _Float16* WkT   = (_Float16*)ws; ws += WSZ;
    _Float16* WvT   = (_Float16*)ws; ws += WSZ;
    _Float16* WoT   = (_Float16*)ws; ws += WSZ;
    _Float16* h_h   = (_Float16*)ws; ws += HSZ;
    _Float16* Qh    = (_Float16*)ws; ws += HSZ;
    _Float16* Kh    = (_Float16*)ws; ws += HSZ;
    _Float16* Vt    = (_Float16*)ws; ws += HSZ;
    _Float16* heads = (_Float16*)ws; ws += HSZ;

    const long prep_total = 4L * (H_ * E_ * KD_) + (long)B_ * N_ * E_;
    prep_kernel<<<(unsigned)((prep_total + 255) / 256), 256, 0, stream>>>(
        h, Wq, Wk, Wv, Wo, WqT, WkT, WvT, WoT, h_h);

    qkv_kernel<<<dim3(B_ * N_ / 16, 3, 1), 32, 0, stream>>>(h_h, WqT, WkT, WvT, Qh, Kh, Vt);

    attn_kernel<<<dim3(N_ / 16, H_, B_), 32, 0, stream>>>(adj, Qh, Kh, Vt, heads);

    outproj_kernel<<<B_ * N_ / 16, 32, 0, stream>>>(heads, WoT, h, out);
}
